// SparseMLP_12919261626601
// MI455X (gfx1250) — matrix operand order repaired and checked by reading, not compiled, not run on hardware
//
#include <hip/hip_runtime.h>

typedef __attribute__((ext_vector_type(16))) __bf16 v16bf;
typedef __attribute__((ext_vector_type(8)))  __bf16 v8bf;
typedef __attribute__((ext_vector_type(4)))  __bf16 v4bf;
typedef __attribute__((ext_vector_type(8)))  float  v8f;
typedef __attribute__((ext_vector_type(4)))  float  v4f;

#define NWEXP 8      // experts (W)
#define NROW  2048   // BHS*SEQ rows per expert
#define HDIM  2048   // hidden size
#define FDIM  1024   // FFN_PER
#define TM    32     // row tile per workgroup
#define KC    32     // K chunk (= WMMA K)
#define NB    256    // output-column panel width (4 waves x 64 cols)
#define BSTR  40     // padded row stride (elements) of transposed B staging (80B, 16B aligned)
#define BBUF  (NB * BSTR)   // elements per staging buffer
#define CHNK  8      // (KC*NB/4)/256 float4 loads per thread per chunk

__device__ __forceinline__ __bf16 f2bf(float f) { return (__bf16)f; }

__device__ __forceinline__ v16bf joinbf(v8bf lo, v8bf hi) {
  return __builtin_shufflevector(lo, hi, 0,1,2,3,4,5,6,7,8,9,10,11,12,13,14,15);
}

__device__ __forceinline__ float gelu_exact(float v) {
  return 0.5f * v * (1.0f + erff(v * 0.70710678118654752f));
}

struct ChunkRegs { v4f v[CHNK]; };

// load a KC x NB fp32 weight chunk into registers (coalesced b128 loads)
__device__ __forceinline__ void load_chunk_regs(const float* __restrict__ g, int ld,
                                                int k0, int n0, int tid, ChunkRegs& r,
                                                bool pf) {
  #pragma unroll
  for (int i = 0; i < CHNK; ++i) {
    int idx = tid + i * 256;
    int k  = idx >> 6;        // 0..31 (NB/4 = 64 float4 per row)
    int c4 = idx & 63;
    const float* p = g + (size_t)(k0 + k) * ld + n0 + c4 * 4;
    r.v[i] = *(const v4f*)p;
    if (pf) __builtin_prefetch(p + (size_t)KC * ld, 0, 1);  // two chunks ahead
  }
}

// convert + store chunk transposed [n][k] as bf16 into an LDS staging buffer
__device__ __forceinline__ void store_chunk_lds(__bf16* buf, int tid,
                                                const ChunkRegs& r) {
  #pragma unroll
  for (int i = 0; i < CHNK; ++i) {
    int idx = tid + i * 256;
    int k  = idx >> 6;
    int c4 = idx & 63;
    v4bf b = __builtin_convertvector(r.v[i], v4bf);
    int nn = c4 * 4;
    buf[(nn + 0) * BSTR + k] = b[0];
    buf[(nn + 1) * BSTR + k] = b[1];
    buf[(nn + 2) * BSTR + k] = b[2];
    buf[(nn + 3) * BSTR + k] = b[3];
  }
}

// one K-step: A fragment from arow, 4 B fragments from buf, 4 WMMAs
__device__ __forceinline__ void mma_step(const __bf16* arow, const __bf16* buf,
                                         int cg, int ln, int hi, v8f acc[4]) {
  v16bf af = joinbf(*(const v8bf*)(arow + 8 * hi), *(const v8bf*)(arow + 16 + 8 * hi));
  #pragma unroll
  for (int tc = 0; tc < 4; ++tc) {
    const __bf16* br = buf + (size_t)(cg * 64 + tc * 16 + ln) * BSTR + 8 * hi;
    v16bf bfv = joinbf(*(const v8bf*)br, *(const v8bf*)(br + 16));
    acc[tc] = __builtin_amdgcn_wmma_f32_16x16x32_bf16(false, af, false, bfv,
                                                      (short)0, acc[tc], false, false);
  }
}

__global__ __launch_bounds__(256)
void fused_mlp_kernel(const float* __restrict__ X,  const float* __restrict__ W1,
                      const float* __restrict__ B1, const float* __restrict__ W2,
                      const float* __restrict__ B2, float* __restrict__ OUT) {
  extern __shared__ char smem[];
  __bf16* xT    = (__bf16*)smem;              // TM x HDIM      (128 KB)
  __bf16* inter = xT + TM * HDIM;             // TM x FDIM      ( 64 KB)
  __bf16* sBt   = inter + TM * FDIM;          // 2 x NB x BSTR  ( 40 KB), transposed
  // double buffer selected by OFFSET on a single LDS base pointer so the
  // address-space stays provably LDS (ds_* ops, not flat_*)

  const int tid  = threadIdx.x;
  const int lane = tid & 31;
  const int wv   = tid >> 5;      // wave 0..7
  const int r    = wv & 1;        // row block (16 rows each)
  const int cg   = wv >> 1;       // column group 0..3 (64 cols each)
  const int ln   = lane & 15;
  const int hi   = lane >> 4;

  const int we   = blockIdx.y;
  const int row0 = blockIdx.x * TM;

  const float* xg  = X  + ((size_t)we * NROW + row0) * HDIM;
  const float* w1g = W1 + (size_t)we * HDIM * FDIM;
  const float* b1g = B1 + (size_t)we * FDIM;
  const float* w2g = W2 + (size_t)we * FDIM * HDIM;
  const float* b2g = B2 + (size_t)we * HDIM;
  float*       outg = OUT + ((size_t)we * NROW + row0) * HDIM;

  // ---- stage x tile (TM x HDIM, contiguous) as bf16 into LDS ----
  {
    const v4f* xg4 = (const v4f*)xg;
    #pragma unroll 4
    for (int i = 0; i < (TM * HDIM / 4) / 256; ++i) {
      int idx = tid + i * 256;
      v4f v = xg4[idx];
      *(v4bf*)(xT + idx * 4) = __builtin_convertvector(v, v4bf);
    }
  }
  __syncthreads();

  const int mrow = 16 * r + ln;   // A-fragment row within tile

  // ===================== Phase 1: inter = gelu(x @ w1 + b1) =====================
  {
    const int NK = HDIM / KC;     // 64
    for (int p = 0; p < FDIM / NB; ++p) {
      const int n0 = p * NB;
      v8f acc[4]; acc[0] = {}; acc[1] = {}; acc[2] = {}; acc[3] = {};
      ChunkRegs rg;
      load_chunk_regs(w1g, FDIM, 0, n0, tid, rg, true);
      store_chunk_lds(sBt, tid, rg);
      __syncthreads();
      for (int kc = 0; kc < NK; ++kc) {
        if (kc + 1 < NK)
          load_chunk_regs(w1g, FDIM, (kc + 1) * KC, n0, tid, rg, kc + 2 < NK);
        mma_step(xT + (size_t)mrow * HDIM + kc * KC, sBt + (kc & 1) * BBUF,
                 cg, ln, hi, acc);
        if (kc + 1 < NK)
          store_chunk_lds(sBt + ((kc + 1) & 1) * BBUF, tid, rg);
        __syncthreads();
      }
      // epilogue: +b1, exact GELU, write bf16 to inter
      #pragma unroll
      for (int tc = 0; tc < 4; ++tc) {
        int nl = n0 + cg * 64 + tc * 16 + ln;
        float bias = b1g[nl];
        #pragma unroll
        for (int i = 0; i < 8; ++i) {
          int m = 16 * r + i + 8 * hi;
          inter[(size_t)m * FDIM + nl] = f2bf(gelu_exact(acc[tc][i] + bias));
        }
      }
      __syncthreads();
    }
  }

  // ===================== Phase 2: out = inter @ w2 + b2 =====================
  {
    const int NK = FDIM / KC;     // 32
    for (int q = 0; q < HDIM / NB; ++q) {
      const int n0 = q * NB;
      v8f acc[4]; acc[0] = {}; acc[1] = {}; acc[2] = {}; acc[3] = {};
      ChunkRegs rg;
      load_chunk_regs(w2g, HDIM, 0, n0, tid, rg, true);
      store_chunk_lds(sBt, tid, rg);
      __syncthreads();
      for (int kc = 0; kc < NK; ++kc) {
        if (kc + 1 < NK)
          load_chunk_regs(w2g, HDIM, (kc + 1) * KC, n0, tid, rg, kc + 2 < NK);
        mma_step(inter + (size_t)mrow * FDIM + kc * KC, sBt + (kc & 1) * BBUF,
                 cg, ln, hi, acc);
        if (kc + 1 < NK)
          store_chunk_lds(sBt + ((kc + 1) & 1) * BBUF, tid, rg);
        __syncthreads();
      }
      // epilogue: +b2, store fp32 to global
      #pragma unroll
      for (int tc = 0; tc < 4; ++tc) {
        int nl = n0 + cg * 64 + tc * 16 + ln;
        float bias = b2g[nl];
        #pragma unroll
        for (int i = 0; i < 8; ++i) {
          int m = 16 * r + i + 8 * hi;
          outg[(size_t)m * HDIM + nl] = acc[tc][i] + bias;
        }
      }
      __syncthreads();
    }
  }
}

// second tuple element: out_bias = b2[:, None, :]  (W x 1 x H)
__global__ void bias_copy_kernel(const float* __restrict__ B2, float* __restrict__ OB) {
  int i = blockIdx.x * blockDim.x + threadIdx.x;
  if (i < NWEXP * HDIM) OB[i] = B2[i];
}

extern "C" void kernel_launch(void* const* d_in, const int* in_sizes, int n_in,
                              void* d_out, int out_size, void* d_ws, size_t ws_size,
                              hipStream_t stream) {
  (void)in_sizes; (void)n_in; (void)d_ws; (void)ws_size; (void)out_size;
  const float* X  = (const float*)d_in[0];
  const float* W1 = (const float*)d_in[1];
  const float* B1 = (const float*)d_in[2];
  const float* W2 = (const float*)d_in[3];
  const float* B2 = (const float*)d_in[4];
  // d_in[5] = idx_list (identity permutation; unused)

  float* out      = (float*)d_out;
  float* out_bias = out + (size_t)NWEXP * NROW * HDIM;

  size_t lds_bytes = (size_t)(TM * HDIM + TM * FDIM + 2 * BBUF) * sizeof(unsigned short);

  dim3 grid(NROW / TM, NWEXP);
  fused_mlp_kernel<<<grid, 256, lds_bytes, stream>>>(X, W1, B1, W2, B2, out);

  bias_copy_kernel<<<(NWEXP * HDIM + 255) / 256, 256, 0, stream>>>(B2, out_bias);
}
